// Decoder_68324339745154
// MI455X (gfx1250) — matrix operand order repaired and checked by reading, not compiled, not run on hardware
//
#include <hip/hip_runtime.h>
#include <hip/hip_bf16.h>
#include <math.h>

// Problem constants (match reference)
#define BB   128
#define TT   64
#define SS   80
#define HH   512
#define EE   512

typedef __bf16 bf16;
typedef __attribute__((ext_vector_type(16))) __bf16 v16bf;
typedef __attribute__((ext_vector_type(8)))  float  v8f;

// Max K across all GEMMs = 1536 -> 48 KB A-tile staging buffer
#define MAX_TILES_K 48

// ---------------------------------------------------------------------------
// Weight packing: f32 row-major [N,K0]|[N,K1] -> bf16 WMMA B-fragment order.
// B-fragment order used here (contiguous k, not AMD's table order): lane L holds column N = L%16;
// half e (0..15) holds K = 16*(L/16) + e.  Tile (tn,tk) stored contiguously:
// dst[((tn*(K/32)+tk)*512) + L*16 + e]
// ---------------------------------------------------------------------------
__global__ __launch_bounds__(256) void pack_b_kernel(
    const float* __restrict__ S0, int K0,
    const float* __restrict__ S1, int K1,
    bf16* __restrict__ dst, int N)
{
    const int K = K0 + K1;
    const int tiles_k = K >> 5;
    const long long total = (long long)N * K;
    long long idx = (long long)blockIdx.x * 256 + threadIdx.x;
    if (idx >= total) return;
    int e    = (int)(idx & 15);
    int Ln   = (int)((idx >> 4) & 31);
    long long tile = idx >> 9;
    int tk = (int)(tile % tiles_k);
    int tn = (int)(tile / tiles_k);
    int g  = Ln >> 4;
    int n  = tn * 16 + (Ln & 15);
    int k  = tk * 32 + 16 * g + e;             // B-fragment K mapping
    float v = (k < K0) ? S0[(long long)n * K0 + k]
                       : S1[(long long)n * K1 + (k - K0)];
    dst[idx] = (bf16)v;
}

// ---------------------------------------------------------------------------
// GEMM: D[128,N] (f32) = A[128,K] (bf16, row-major, lda) x Bpacked (fragments)
// Block = 256 threads = 8 waves. The FULL 16xK A tile for this block's m-tile
// is staged once into LDS in A-fragment order (one barrier). Each wave owns
// TWO 16x16 N tiles (two independent accumulator chains sharing each A
// fragment), and the K loop is software-pipelined: fragments for kt+1 are
// loaded before the WMMAs for kt, so waits overlap with the matrix pipe.
// A-fragment order used here (contiguous k, same as B): lane L row M=L%16, half e -> K:
//   K = 16*(L/16)+e
// ---------------------------------------------------------------------------
__global__ __launch_bounds__(256) void gemm_bf16_wmma(
    const bf16* __restrict__ A, int lda,
    const bf16* __restrict__ Bp,
    float* __restrict__ D, int N, int K)
{
    __shared__ __align__(32) bf16 asmem[MAX_TILES_K * 512];   // 48 KB

    const int tid     = threadIdx.x;
    const int lane    = tid & 31;
    const int wave    = tid >> 5;
    const int m_tile  = blockIdx.x & 7;           // M = 128 -> 8 tiles
    const int n_tile0 = (blockIdx.x >> 3) * 16 + wave;
    const int n_tile1 = n_tile0 + 8;
    const int m0      = m_tile * 16;
    const int tiles_k = K >> 5;

    // Stage entire 16xK A tile in fragment order.
    // Thread t loads A[m0+sm][kt*32+kc .. +1] (coalesced b32) and stores it at
    // the A-fragment LDS slot (same lane L, adjacent halves e,e+1).
    const int sm = tid >> 4;                      // 0..15 row
    const int kc = (tid & 15) << 1;               // 0,2,..,30 col pair
    int g, e;
    g = kc >> 4; e = kc & 15;
    const int sfrag = (((g << 4) | sm) << 4) + e; // L*16 + e (e even)

    const bf16* __restrict__ arow = A + (long long)(m0 + sm) * lda + kc;
#pragma unroll 4
    for (int kt = 0; kt < tiles_k; ++kt) {
        unsigned int pair =
            *reinterpret_cast<const unsigned int*>(arow + kt * 32);
        *reinterpret_cast<unsigned int*>(&asmem[kt * 512 + sfrag]) = pair;
    }
    __syncthreads();

    v8f acc0 = {0.f, 0.f, 0.f, 0.f, 0.f, 0.f, 0.f, 0.f};
    v8f acc1 = {0.f, 0.f, 0.f, 0.f, 0.f, 0.f, 0.f, 0.f};
    const bf16* __restrict__ bb0 =
        Bp + ((long long)n_tile0 * tiles_k) * 512 + lane * 16;
    const bf16* __restrict__ bb1 =
        Bp + ((long long)n_tile1 * tiles_k) * 512 + lane * 16;
    const bf16* __restrict__ afrag = asmem + lane * 16;

    // Software pipeline: prologue loads for kt=0
    v16bf a_cur  = *reinterpret_cast<const v16bf*>(afrag);
    v16bf b0_cur = *reinterpret_cast<const v16bf*>(bb0);
    v16bf b1_cur = *reinterpret_cast<const v16bf*>(bb1);

#pragma unroll 4
    for (int kt = 0; kt < tiles_k - 1; ++kt) {
        v16bf a_nxt  = *reinterpret_cast<const v16bf*>(afrag + (kt + 1) * 512);
        v16bf b0_nxt = *reinterpret_cast<const v16bf*>(bb0 + (long long)(kt + 1) * 512);
        v16bf b1_nxt = *reinterpret_cast<const v16bf*>(bb1 + (long long)(kt + 1) * 512);
        __builtin_prefetch(bb0 + (long long)(kt + 2) * 512, 0, 3);
        __builtin_prefetch(bb1 + (long long)(kt + 2) * 512, 0, 3);
        acc0 = __builtin_amdgcn_wmma_f32_16x16x32_bf16(
            false, a_cur, false, b0_cur, (short)0, acc0, false, false);
        acc1 = __builtin_amdgcn_wmma_f32_16x16x32_bf16(
            false, a_cur, false, b1_cur, (short)0, acc1, false, false);
        a_cur = a_nxt; b0_cur = b0_nxt; b1_cur = b1_nxt;
    }
    acc0 = __builtin_amdgcn_wmma_f32_16x16x32_bf16(
        false, a_cur, false, b0_cur, (short)0, acc0, false, false);
    acc1 = __builtin_amdgcn_wmma_f32_16x16x32_bf16(
        false, a_cur, false, b1_cur, (short)0, acc1, false, false);

    const int mrow  = m0 + ((lane >> 4) << 3);
    const int ncol0 = n_tile0 * 16 + (lane & 15);
    const int ncol1 = n_tile1 * 16 + (lane & 15);
#pragma unroll
    for (int r = 0; r < 8; ++r) {
        D[(long long)(mrow + r) * N + ncol0] = acc0[r];
        D[(long long)(mrow + r) * N + ncol1] = acc1[r];
    }
}

// ---------------------------------------------------------------------------
// Build step inputs: xh0[b, 0:512]=emb(token), [512:1024]=out_prev (input
// feeding), [1024:1536]=h_prev(layer0); xh1[b, 512:1024]=h_prev(layer1).
// ---------------------------------------------------------------------------
__global__ __launch_bounds__(256) void build_inputs_kernel(
    const int*   __restrict__ tokens_t,
    const float* __restrict__ emb,
    const float* __restrict__ out_prev,
    const float* __restrict__ h0_prev,
    const float* __restrict__ h1_prev,
    bf16* __restrict__ xh0, bf16* __restrict__ xh1)
{
    int idx = blockIdx.x * 256 + threadIdx.x;    // 128*2048 threads
    int b = idx >> 11;
    int c = idx & 2047;
    if (c < 1536) {
        float v;
        if (c < 512)       v = emb[(long long)tokens_t[b] * EE + c];
        else if (c < 1024) v = out_prev[b * HH + (c - 512)];
        else               v = h0_prev[b * HH + (c - 1024)];
        xh0[(long long)b * 1536 + c] = (bf16)v;
    } else {
        int j = c - 1536;
        xh1[(long long)b * 1024 + 512 + j] = (bf16)h1_prev[b * HH + j];
    }
}

// ---------------------------------------------------------------------------
// LSTM pointwise: gates [128,2048] (i|f|g|o) + biases -> new c, h.
// Writes f32 state and a bf16 copy of h at xhdst[b*ld + off + j].
// ---------------------------------------------------------------------------
__global__ __launch_bounds__(256) void lstm_pointwise_kernel(
    const float* __restrict__ gates,
    const float* __restrict__ bih, const float* __restrict__ bhh,
    const float* __restrict__ c_prev,
    float* __restrict__ c_out, float* __restrict__ h_out,
    bf16* __restrict__ xhdst, int ld, int off)
{
    int idx = blockIdx.x * 256 + threadIdx.x;    // 128*512 threads
    int b = idx >> 9, j = idx & 511;
    const float* gr = gates + (long long)b * 2048;
    float gi = gr[j]          + bih[j]          + bhh[j];
    float gf = gr[512 + j]    + bih[512 + j]    + bhh[512 + j];
    float gg = gr[1024 + j]   + bih[1024 + j]   + bhh[1024 + j];
    float go = gr[1536 + j]   + bih[1536 + j]   + bhh[1536 + j];
    float i = 1.f / (1.f + __expf(-gi));
    float f = 1.f / (1.f + __expf(-gf));
    float o = 1.f / (1.f + __expf(-go));
    float gtan = tanhf(gg);
    float cn = f * c_prev[idx] + i * gtan;
    float hn = o * tanhf(cn);
    c_out[idx] = cn;
    h_out[idx] = hn;
    xhdst[(long long)b * ld + off + j] = (bf16)hn;
}

// ---------------------------------------------------------------------------
// Attention: one block (8 waves) per batch row.
// scores: each wave handles 10 of the S=80 rows, lanes sum 16 elements each
// (coalesced), then shuffle-reduce. softmax by thread 0 (S=80 is tiny).
// wctx -> xa[b, 0:512] (bf16); attn probabilities -> d_out.
// ---------------------------------------------------------------------------
__global__ __launch_bounds__(256) void attention_kernel(
    const float* __restrict__ q,
    const float* __restrict__ context,
    float* __restrict__ attn_out,
    bf16* __restrict__ xa)
{
    const int b    = blockIdx.x;
    const int tid  = threadIdx.x;
    const int lane = tid & 31;
    const int wave = tid >> 5;
    __shared__ float qs[HH];
    __shared__ float sc[SS];
    __shared__ float invsum;

    qs[tid]       = q[b * HH + tid];
    qs[tid + 256] = q[b * HH + 256 + tid];
    __syncthreads();

    for (int s = wave; s < SS; s += 8) {
        const float* crow = context + ((long long)b * SS + s) * HH;
        float acc = 0.f;
#pragma unroll
        for (int i = 0; i < HH / 32; ++i)
            acc += crow[i * 32 + lane] * qs[i * 32 + lane];
#pragma unroll
        for (int off = 16; off > 0; off >>= 1)
            acc += __shfl_down(acc, off, 32);
        if (lane == 0) sc[s] = acc;
    }
    __syncthreads();

    if (tid == 0) {
        float mx = sc[0];
        for (int s = 1; s < SS; ++s) mx = fmaxf(mx, sc[s]);
        float sum = 0.f;
        for (int s = 0; s < SS; ++s) { float ev = __expf(sc[s] - mx); sc[s] = ev; sum += ev; }
        invsum = 1.f / sum;
    }
    __syncthreads();

    if (tid < SS) {
        float a = sc[tid] * invsum;
        sc[tid] = a;
        attn_out[b * SS + tid] = a;
    }
    __syncthreads();

    for (int h = tid; h < HH; h += 256) {
        float acc = 0.f;
#pragma unroll 8
        for (int s = 0; s < SS; ++s)
            acc += sc[s] * context[((long long)b * SS + s) * HH + h];
        xa[(long long)b * 1024 + h] = (bf16)acc;
    }
}

// out = tanh(pre); write timestep output and input-feeding buffer
__global__ __launch_bounds__(256) void out_tanh_kernel(
    const float* __restrict__ pre,
    float* __restrict__ out_t, float* __restrict__ out_prev)
{
    int idx = blockIdx.x * 256 + threadIdx.x;    // 128*512
    float v = tanhf(pre[idx]);
    out_t[idx]    = v;
    out_prev[idx] = v;
}

__global__ __launch_bounds__(256) void zero_kernel(float* __restrict__ p, int n)
{
    int idx = blockIdx.x * 256 + threadIdx.x;
    if (idx < n) p[idx] = 0.f;
}

// ---------------------------------------------------------------------------
extern "C" void kernel_launch(void* const* d_in, const int* in_sizes, int n_in,
                              void* d_out, int out_size, void* d_ws, size_t ws_size,
                              hipStream_t stream)
{
    const int*   tokens  = (const int*)  d_in[0];   // [T,B]
    const float* h0_in   = (const float*)d_in[1];   // [L,B,H]
    const float* c0_in   = (const float*)d_in[2];   // [L,B,H]
    const float* context = (const float*)d_in[3];   // [B,S,H]
    const float* emb     = (const float*)d_in[4];   // [V,E]
    const float* Wih0    = (const float*)d_in[5];   // [2048,1024]
    const float* Whh0    = (const float*)d_in[6];   // [2048,512]
    const float* bih0    = (const float*)d_in[7];
    const float* bhh0    = (const float*)d_in[8];
    const float* Wih1    = (const float*)d_in[9];   // [2048,512]
    const float* Whh1    = (const float*)d_in[10];  // [2048,512]
    const float* bih1    = (const float*)d_in[11];
    const float* bhh1    = (const float*)d_in[12];
    const float* W_in    = (const float*)d_in[13];  // [512,512]
    const float* W_out   = (const float*)d_in[14];  // [512,1024]

    // d_out layout: outputs [T,B,H] | hT [L,B,H] | cT [L,B,H] | attns [T,B,S]
    float* outputs = (float*)d_out;
    float* hT      = outputs + (long long)TT * BB * HH;
    float* cT      = hT + 2LL * BB * HH;
    float* attns   = cT + 2LL * BB * HH;

    // Workspace carve-up (256B aligned regions)
    char* p = (char*)d_ws;
    auto carve = [&](size_t bytes) -> void* {
        void* r = (void*)p;
        p += (bytes + 255) & ~(size_t)255;
        return r;
    };
    bf16*  W0p      = (bf16*) carve(2048LL * 1536 * 2);  // [Wih0|Whh0] packed
    bf16*  W1p      = (bf16*) carve(2048LL * 1024 * 2);  // [Wih1|Whh1] packed
    bf16*  Wqp      = (bf16*) carve(512LL * 512 * 2);    // W_in packed
    bf16*  Wop      = (bf16*) carve(512LL * 1024 * 2);   // W_out packed
    bf16*  xh0      = (bf16*) carve(128LL * 1536 * 2);   // [emb|out_prev|h0]
    bf16*  xh1      = (bf16*) carve(128LL * 1024 * 2);   // [h0_new|h1_prev]
    bf16*  xa       = (bf16*) carve(128LL * 1024 * 2);   // [wctx|rnn_out]
    float* gates    = (float*)carve(128LL * 2048 * 4);
    float* qbuf     = (float*)carve(128LL * 512 * 4);
    float* outpre   = (float*)carve(128LL * 512 * 4);
    float* out_prev = (float*)carve(128LL * 512 * 4);

    // One-time (per launch) weight packing into WMMA fragment order
    pack_b_kernel<<<(2048 * 1536) / 256, 256, 0, stream>>>(Wih0, 1024, Whh0, 512, W0p, 2048);
    pack_b_kernel<<<(2048 * 1024) / 256, 256, 0, stream>>>(Wih1,  512, Whh1, 512, W1p, 2048);
    pack_b_kernel<<<(512  * 512 ) / 256, 256, 0, stream>>>(W_in,  512, (const float*)nullptr, 0, Wqp, 512);
    pack_b_kernel<<<(512  * 1024) / 256, 256, 0, stream>>>(W_out, 1024, (const float*)nullptr, 0, Wop, 512);
    zero_kernel<<<(BB * HH + 255) / 256, 256, 0, stream>>>(out_prev, BB * HH);

    for (int t = 0; t < TT; ++t) {
        const float* h_prev = (t == 0) ? h0_in : hT;
        const float* c_prev = (t == 0) ? c0_in : cT;

        // Input feeding + embedding + h_prev staging (bf16)
        build_inputs_kernel<<<(128 * 2048) / 256, 256, 0, stream>>>(
            tokens + t * BB, emb, out_prev,
            h_prev, h_prev + BB * HH, xh0, xh1);

        // Layer 0 gates: [128,1536] x [1536,2048]
        gemm_bf16_wmma<<<8 * (2048 / 256), 256, 0, stream>>>(
            xh0, 1536, W0p, gates, 2048, 1536);
        lstm_pointwise_kernel<<<(BB * HH) / 256, 256, 0, stream>>>(
            gates, bih0, bhh0, c_prev, cT, hT, xh1, 1024, 0);

        // Layer 1 gates: [128,1024] x [1024,2048]
        gemm_bf16_wmma<<<8 * (2048 / 256), 256, 0, stream>>>(
            xh1, 1024, W1p, gates, 2048, 1024);
        lstm_pointwise_kernel<<<(BB * HH) / 256, 256, 0, stream>>>(
            gates, bih1, bhh1, c_prev + BB * HH,
            cT + BB * HH, hT + BB * HH, xa, 1024, 512);

        // q = rnn_out @ W_in^T : A = xa[:,512:1024] (lda=1024)
        gemm_bf16_wmma<<<8 * (512 / 256), 256, 0, stream>>>(
            xa + 512, 1024, Wqp, qbuf, 512, 512);

        // softmax(context.q), weighted context -> xa[:,0:512]
        attention_kernel<<<BB, 256, 0, stream>>>(
            qbuf, context, attns + (long long)t * BB * SS, xa);

        // out = tanh([wctx|rnn_out] @ W_out^T)
        gemm_bf16_wmma<<<8 * (512 / 256), 256, 0, stream>>>(
            xa, 1024, Wop, outpre, 512, 1024);
        out_tanh_kernel<<<(BB * HH) / 256, 256, 0, stream>>>(
            outpre, outputs + (long long)t * BB * HH, out_prev);
    }
}
